// LinkPrediction_62586263437676
// MI455X (gfx1250) — compile-verified, matrix operand order checked by reading, not run on hardware
//
#include <hip/hip_runtime.h>
#include <stdint.h>

// ---------------------------------------------------------------------------
// CDNA5 (gfx1250) bf16 WMMA GEMM pipeline for the LinkPrediction reference.
// Heavy math: v_wmma_f32_16x16x32_bf16 (f32 accumulate).
// Tile staging: global_load_async_to_lds_b128 (ASYNCcnt path), with
// synchronous b128+ds fallback if the builtin is unavailable.
// ---------------------------------------------------------------------------

typedef __attribute__((ext_vector_type(16))) __bf16 v16bf;
typedef __attribute__((ext_vector_type(8)))  float  v8f;
typedef __attribute__((ext_vector_type(4)))  int    v4i;

typedef __attribute__((address_space(1))) v4i* gptr_v4i;
typedef __attribute__((address_space(3))) v4i* lptr_v4i;

#define TILE_M 128
#define TILE_N 128
#define TILE_K 32
#define LDSS   56   // row stride (elems): 112B rows -> 16B aligned, 28-bank rotation (conflict-free)

#if __has_builtin(__builtin_amdgcn_global_load_async_to_lds_b128)
#define USE_ASYNC_LDS 1
#else
#define USE_ASYNC_LDS 0
#endif

__device__ __forceinline__ void wait_asynccnt0() {
#if __has_builtin(__builtin_amdgcn_s_wait_asynccnt)
    __builtin_amdgcn_s_wait_asynccnt(0);
#else
    asm volatile("s_wait_asynccnt 0" ::: "memory");
#endif
}

// copy 8 bf16 (16B) global -> LDS
__device__ __forceinline__ void copy16(const __bf16* g, __bf16* l) {
#if USE_ASYNC_LDS
    __builtin_amdgcn_global_load_async_to_lds_b128(
        (gptr_v4i)(void*)const_cast<__bf16*>(g),
        (lptr_v4i)(void*)l, 0, 0);
#else
    uint4 d = *(const uint4*)g;
    *(uint2*)(l)     = make_uint2(d.x, d.y);
    *(uint2*)(l + 4) = make_uint2(d.z, d.w);
#endif
}

union Frag16 { v16bf v; uint2 u[4]; };

// f32 [K,Ncols] -> bf16 transposed [Ncols,K], LDS-tiled (coalesced both sides)
__global__ __launch_bounds__(256) void cvt_transpose_bf16(
    const float* __restrict__ W, __bf16* __restrict__ Wt, int K, int Ncols)
{
    __shared__ float tile[32][33];
    int kb = blockIdx.x * 32, nb = blockIdx.y * 32;
    int tx = threadIdx.x & 31, ty = threadIdx.x >> 5;   // ty 0..7
    #pragma unroll
    for (int i = 0; i < 32; i += 8)
        tile[ty + i][tx] = W[(size_t)(kb + ty + i) * Ncols + nb + tx];
    __syncthreads();
    #pragma unroll
    for (int i = 0; i < 32; i += 8)
        Wt[(size_t)(nb + ty + i) * K + kb + tx] = (__bf16)tile[tx][ty + i];
}

// AMODE: 0 = A is f32 (row stride lda), 1 = A is bf16, 2 = gathered concat A
//   (cols 0..767 emb[src], 768..1535 emb[dst], 1536..2303 yproj[batch[src]])
// ACT: 0 none, 1 relu, 2 leaky_relu(0.4)   (compile-time: branch-free epilogue)
template<int AMODE, int ACT>
__global__ __launch_bounds__(256) void gemm_bf16_wmma(
    const void* __restrict__ Aptr, long lda,
    const __bf16* __restrict__ Bt,     // [Ncols, K] bf16, PRE-TRANSPOSED
    const float* __restrict__ bias,    // [Ncols]
    __bf16* __restrict__ out, long ldo,
    int M, int K, int Ncols,
    const int* __restrict__ eidx, const int* __restrict__ batch,
    const __bf16* __restrict__ yproj, int E)
{
    __shared__ __bf16 lA[TILE_M * LDSS];
    __shared__ __bf16 lB[TILE_N * LDSS];   // lB[n*LDSS + k]

    const int t     = threadIdx.x;
    const int lane  = t & 31;
    const int wid   = t >> 5;
    const int wm    = wid >> 1;            // 0..3 : 32-row strip
    const int wn    = wid & 1;             // 0..1 : 64-col strip
    const int l16   = lane & 15;
    const int khalf = (lane >> 4) * 8;     // K-half select (A/B frag layout)
    const int mhalf = (lane >> 4) * 8;     // M-half select (C/D layout)

    // staging coords: 16B chunk per thread, 128 rows x 4 chunks, 2 iters
    const int r0 = t >> 2;        // 0..63
    const int c8 = (t & 3) * 8;   // 0,8,16,24

    v8f acc[2][4] = {};
    const int kTiles = K / TILE_K;

    for (int kt = 0; kt < kTiles; ++kt) {
        const int k0 = kt * TILE_K;

        // ---- stage A tile (128 x 32 bf16); rows clamped (dup rows never stored) ----
        if (AMODE == 0) {
            const float* A = (const float*)Aptr;
            int ar0 = t >> 3;
            int ac4 = (t & 7) * 4;
            #pragma unroll
            for (int i = 0; i < 4; ++i) {
                int r   = ar0 + i * 32;
                int gr  = blockIdx.x * TILE_M + r;
                int grc = gr < M ? gr : M - 1;
                float4 f = *(const float4*)(A + (size_t)grc * lda + k0 + ac4);
                __bf16 tmp[4] = {(__bf16)f.x, (__bf16)f.y, (__bf16)f.z, (__bf16)f.w};
                *(uint2*)&lA[r * LDSS + ac4] = *(uint2*)tmp;
            }
        } else {
            #pragma unroll
            for (int i = 0; i < 2; ++i) {
                int r   = r0 + i * 64;
                int gr  = blockIdx.x * TILE_M + r;
                int grc = gr < M ? gr : M - 1;
                const __bf16* src;
                if (AMODE == 1) {
                    src = (const __bf16*)Aptr + (size_t)grc * lda + k0 + c8;
                } else {
                    const int seg  = k0 / 768;            // uniform per K-tile
                    const int koff = k0 - seg * 768;
                    int s = eidx[grc];
                    if (seg == 0)
                        src = (const __bf16*)Aptr + (size_t)s * 768 + koff + c8;
                    else if (seg == 1)
                        src = (const __bf16*)Aptr + (size_t)eidx[E + grc] * 768 + koff + c8;
                    else
                        src = yproj + (size_t)batch[s] * 768 + koff + c8;
                }
                copy16(src, &lA[r * LDSS + c8]);
            }
        }

        // ---- stage B tile: rows of pre-transposed Wt, straight copy ----
        #pragma unroll
        for (int i = 0; i < 2; ++i) {
            int n = r0 + i * 64;
            const __bf16* src = Bt + (size_t)(blockIdx.y * TILE_N + n) * K + k0 + c8;
            copy16(src, &lB[n * LDSS + c8]);
        }

#if USE_ASYNC_LDS
        wait_asynccnt0();
#endif
        __syncthreads();

        // ---- fragments + WMMA (EXEC all-ones: no divergence here) ----
        Frag16 af[2], bf[4];
        #pragma unroll
        for (int i = 0; i < 2; ++i) {
            const __bf16* p = &lA[(wm * 32 + i * 16 + l16) * LDSS + khalf];
            af[i].u[0] = *(const uint2*)(p);
            af[i].u[1] = *(const uint2*)(p + 4);
            af[i].u[2] = *(const uint2*)(p + 16);
            af[i].u[3] = *(const uint2*)(p + 20);
        }
        #pragma unroll
        for (int j = 0; j < 4; ++j) {
            const __bf16* p = &lB[(wn * 64 + j * 16 + l16) * LDSS + khalf];
            bf[j].u[0] = *(const uint2*)(p);
            bf[j].u[1] = *(const uint2*)(p + 4);
            bf[j].u[2] = *(const uint2*)(p + 16);
            bf[j].u[3] = *(const uint2*)(p + 20);
        }
        #pragma unroll
        for (int i = 0; i < 2; ++i)
            #pragma unroll
            for (int j = 0; j < 4; ++j)
                acc[i][j] = __builtin_amdgcn_wmma_f32_16x16x32_bf16(
                    false, af[i].v, false, bf[j].v, (short)0, acc[i][j], false, false);
        __syncthreads();
    }

    // ---- epilogue: bias + activation (compile-time), store bf16 ----
    #pragma unroll
    for (int i = 0; i < 2; ++i) {
        #pragma unroll
        for (int j = 0; j < 4; ++j) {
            int col = blockIdx.y * TILE_N + wn * 64 + j * 16 + l16;
            float bv = bias[col];
            #pragma unroll
            for (int v = 0; v < 8; ++v) {
                int row = blockIdx.x * TILE_M + wm * 32 + i * 16 + mhalf + v;
                float f = acc[i][j][v] + bv;
                if (ACT == 1)      f = f > 0.f ? f : 0.f;
                else if (ACT == 2) f = f > 0.f ? f : 0.4f * f;
                if (row < M) out[(size_t)row * ldo + col] = (__bf16)f;
            }
        }
    }
}

// Final head: q2[E,768] @ pc2_W[768,2] + b, sigmoid; also copy edge_label.
// One wave32 per row; shuffle reduction.
__global__ __launch_bounds__(256) void pc2_sigmoid_kernel(
    const __bf16* __restrict__ q2, const float* __restrict__ W2,
    const float* __restrict__ b2, const float* __restrict__ label,
    float* __restrict__ out, int E)
{
    int row  = blockIdx.x * 8 + (threadIdx.x >> 5);
    int lane = threadIdx.x & 31;
    if (row >= E) return;
    float s0 = 0.f, s1 = 0.f;
    const __bf16* q = q2 + (size_t)row * 768;
    #pragma unroll
    for (int j = 0; j < 3; ++j) {
        int c0 = lane * 8 + j * 256;
        uint4 d = *(const uint4*)(q + c0);
        __bf16 tmp[8];
        *(uint4*)tmp = d;
        #pragma unroll
        for (int e = 0; e < 8; ++e) {
            float f = (float)tmp[e];
            s0 += f * W2[(c0 + e) * 2 + 0];
            s1 += f * W2[(c0 + e) * 2 + 1];
        }
    }
    #pragma unroll
    for (int off = 16; off > 0; off >>= 1) {
        s0 += __shfl_xor(s0, off, 32);
        s1 += __shfl_xor(s1, off, 32);
    }
    if (lane == 0) {
        out[(size_t)row * 2 + 0] = 1.f / (1.f + __expf(-(s0 + b2[0])));
        out[(size_t)row * 2 + 1] = 1.f / (1.f + __expf(-(s1 + b2[1])));
        out[(size_t)2 * E + row] = label[row];
    }
}

extern "C" void kernel_launch(void* const* d_in, const int* in_sizes, int n_in,
                              void* d_out, int out_size, void* d_ws, size_t ws_size,
                              hipStream_t stream)
{
    (void)in_sizes; (void)n_in; (void)out_size; (void)ws_size;
    const int Nn = 40000, Ee = 100000, Gg = 128;

    const float* x    = (const float*)d_in[0];
    const int*   eidx = (const int*)  d_in[1];
    const int*   batch= (const int*)  d_in[2];
    // d_in[3] edge_attr and d_in[4] train_edge_mask are unused by the math.
    const float* y    = (const float*)d_in[5];
    const float* elab = (const float*)d_in[6];
    const float* ahW  = (const float*)d_in[7];  const float* ahb = (const float*)d_in[8];
    const float* c0W  = (const float*)d_in[9];  const float* c0b = (const float*)d_in[10];
    const float* c1W  = (const float*)d_in[11]; const float* c1b = (const float*)d_in[12];
    const float* c2W  = (const float*)d_in[13]; const float* c2b = (const float*)d_in[14];
    const float* o1W  = (const float*)d_in[15]; const float* o1b = (const float*)d_in[16];
    const float* o2W  = (const float*)d_in[17]; const float* o2b = (const float*)d_in[18];
    const float* pyW  = (const float*)d_in[19]; const float* pyb = (const float*)d_in[20];
    const float* p1W  = (const float*)d_in[21]; const float* p1b = (const float*)d_in[22];
    const float* p2W  = (const float*)d_in[23]; const float* p2b = (const float*)d_in[24];
    float* out = (float*)d_out;

    // ---- workspace layout (~320 MB, 256B aligned slabs) ----
    uint8_t* p = (uint8_t*)d_ws;
    auto alloc_bf = [&](size_t elems) -> __bf16* {
        __bf16* r = (__bf16*)p;
        p += ((elems * sizeof(__bf16) + 255) & ~(size_t)255);
        return r;
    };
    __bf16* wAH  = alloc_bf(768 * 768);      // all weights stored TRANSPOSED [N,K] bf16
    __bf16* wC0  = alloc_bf(768 * 768);
    __bf16* wC1  = alloc_bf(768 * 768);
    __bf16* wC2  = alloc_bf(768 * 768);
    __bf16* wO1  = alloc_bf((size_t)768 * 2304);
    __bf16* wO2  = alloc_bf(768 * 768);
    __bf16* wPY  = alloc_bf(768 * 768);
    __bf16* wP1  = alloc_bf((size_t)768 * 2304);
    __bf16* yprj = alloc_bf((size_t)Gg * 768);
    __bf16* hbuf = alloc_bf((size_t)Nn * 768);   // h, later reused as t
    __bf16* embb = alloc_bf((size_t)Nn * 768);   // final node embeddings
    __bf16* nebf = alloc_bf((size_t)Nn * 2304);  // node_emb concat; reused as q2 [E,768]
    __bf16* q2bf = nebf;                         // E*768 <= Nn*2304

    auto cvtT = [&](const float* s, __bf16* d, int K, int Nc) {
        cvt_transpose_bf16<<<dim3(K / 32, Nc / 32), 256, 0, stream>>>(s, d, K, Nc);
    };
    cvtT(ahW, wAH, 768, 768);
    cvtT(c0W, wC0, 768, 768);
    cvtT(c1W, wC1, 768, 768);
    cvtT(c2W, wC2, 768, 768);
    cvtT(o1W, wO1, 2304, 768);
    cvtT(o2W, wO2, 768, 768);
    cvtT(pyW, wPY, 768, 768);
    cvtT(p1W, wP1, 2304, 768);

    const dim3 blk(256);
    const dim3 gN((Nn + TILE_M - 1) / TILE_M, 768 / TILE_N);
    const dim3 gG((Gg + TILE_M - 1) / TILE_M, 768 / TILE_N);
    const dim3 gE((Ee + TILE_M - 1) / TILE_M, 768 / TILE_N);

    // h = relu(x @ ahW + ahb)
    gemm_bf16_wmma<0, 1><<<gN, blk, 0, stream>>>(x, 768, wAH, ahb, hbuf, 768,
        Nn, 768, 768, nullptr, nullptr, nullptr, 0);
    // h0..h2 written into node_emb slices (ldo = 2304)
    gemm_bf16_wmma<1, 1><<<gN, blk, 0, stream>>>(hbuf, 768, wC0, c0b, nebf + 0,    2304,
        Nn, 768, 768, nullptr, nullptr, nullptr, 0);
    gemm_bf16_wmma<1, 1><<<gN, blk, 0, stream>>>(nebf + 0, 2304, wC1, c1b, nebf + 768,  2304,
        Nn, 768, 768, nullptr, nullptr, nullptr, 0);
    gemm_bf16_wmma<1, 1><<<gN, blk, 0, stream>>>(nebf + 768, 2304, wC2, c2b, nebf + 1536, 2304,
        Nn, 768, 768, nullptr, nullptr, nullptr, 0);
    // t = relu(node_emb @ o1W + o1b)   (reuses hbuf)
    gemm_bf16_wmma<1, 1><<<gN, blk, 0, stream>>>(nebf, 2304, wO1, o1b, hbuf, 768,
        Nn, 2304, 768, nullptr, nullptr, nullptr, 0);
    // emb = t @ o2W + o2b
    gemm_bf16_wmma<1, 0><<<gN, blk, 0, stream>>>(hbuf, 768, wO2, o2b, embb, 768,
        Nn, 768, 768, nullptr, nullptr, nullptr, 0);
    // yproj = leaky_relu(y @ pyW + pyb, 0.4)  — only G=128 rows needed
    gemm_bf16_wmma<0, 2><<<gG, blk, 0, stream>>>(y, 768, wPY, pyb, yprj, 768,
        Gg, 768, 768, nullptr, nullptr, nullptr, 0);
    // q2 = relu(concat(emb[src], emb[dst], yproj[batch[src]]) @ pc1_W + b)
    //      (gather fused into A-tile async loads; writes into recycled nebf)
    gemm_bf16_wmma<2, 1><<<gE, blk, 0, stream>>>(embb, 0, wP1, p1b, q2bf, 768,
        Ee, 2304, 768, eidx, batch, yprj, Ee);
    // pred = sigmoid(q2 @ pc2_W + b); std = edge_label
    pc2_sigmoid_kernel<<<dim3((Ee + 7) / 8), blk, 0, stream>>>(q2bf, p2W, p2b, elab, out, Ee);
}
